// GraphSAGE_17540646436883
// MI455X (gfx1250) — matrix operand order repaired and checked by reading, not compiled, not run on hardware
//
#include <hip/hip_runtime.h>
#include <stdint.h>

#define NN 100000
#define NE 1600000
#define DD 128
#define NL 3
#define MT 32   // rows per block (100000 / 32 = 3125 exact)

typedef __attribute__((ext_vector_type(16))) __bf16 v16bf;
typedef __attribute__((ext_vector_type(8)))  float  v8f;

// ---------------- utility kernels ----------------

__global__ void k_zero_f4(float4* __restrict__ p, int n4) {
  int i = blockIdx.x * blockDim.x + threadIdx.x;
  if (i < n4) p[i] = make_float4(0.f, 0.f, 0.f, 0.f);
}

__global__ void k_zero_f(float* __restrict__ p, int n) {
  int i = blockIdx.x * blockDim.x + threadIdx.x;
  if (i < n) p[i] = 0.f;
}

// count in-degree: tgt points at edge_index[1] (int64)
__global__ void k_count(const long long* __restrict__ tgt, float* __restrict__ cnt) {
  int e = blockIdx.x * blockDim.x + threadIdx.x;
  if (e < NE) atomicAdd(&cnt[(int)tgt[e]], 1.0f);
}

__global__ void k_inv(float* __restrict__ cnt) {
  int i = blockIdx.x * blockDim.x + threadIdx.x;
  if (i < NN) cnt[i] = 1.0f / fmaxf(cnt[i], 1.0f);
}

// Pack W [L][128][128] f32 into WMMA B-fragment order, bf16:
// P[((l*8 + tile)*4 + kk)*32 + lane][e]  =  W[l][32*kk + 16*hi + e][tile*16 + m]
// so a lane's 16-element fragment is 32 contiguous bytes.
__global__ void k_pack_w(const float* __restrict__ W, __bf16* __restrict__ P, int total) {
  int i = blockIdx.x * blockDim.x + threadIdx.x;  // over L*16384
  if (i >= total) return;
  int l    = i >> 14;
  int r    = i & 16383;
  int e    = r & 15;
  int lane = (r >> 4) & 31;
  int kk   = (r >> 9) & 3;
  int tile = r >> 11;
  int m  = lane & 15, hi = lane >> 4;
  int k  = 32 * kk + 16 * hi + e;
  int n  = tile * 16 + m;
  P[i] = (__bf16)W[(size_t)l * DD * DD + k * DD + n];
}

// ---------------- edge scatter: agg[tgt] += hp[src] ----------------
// one wave per edge: 32 lanes x float4 = 512B row; atomics land in L2
__global__ void k_scatter(const long long* __restrict__ ei,
                          const float* __restrict__ hp,
                          float* __restrict__ agg) {
  long long t = (long long)blockIdx.x * blockDim.x + threadIdx.x; // [0, NE*32)
  int e = (int)(t >> 5);
  int c = ((int)t & 31) * 4;
  int src = (int)ei[e];
  int tg  = (int)ei[NE + e];
  const float4 v = *(const float4*)(hp + (size_t)src * DD + c);
  float* dst = agg + (size_t)tg * DD + c;
  atomicAdd(dst + 0, v.x);
  atomicAdd(dst + 1, v.y);
  atomicAdd(dst + 2, v.z);
  atomicAdd(dst + 3, v.w);
}

// ---------------- fragment helpers ----------------

__device__ __forceinline__ v16bf load_a_frag(const __bf16 (*As)[136], int m, int hi, int kk) {
  v16bf a;
  // VGPR j: K = 32kk + 16hi + 2j (matches k_pack_w B layout)
  // -> one contiguous 16-element (32B) run, vectorizes to 2x ds_load_b128
  #pragma unroll
  for (int j = 0; j < 8; ++j) {
    int kb = 32 * kk + 2 * j + 16 * hi;
    a[2 * j]     = As[m][kb];
    a[2 * j + 1] = As[m][kb + 1];
  }
  return a;
}

// ---------------- WMMA GEMM: out = act(A @ W + b) ----------------
// 256 thr = 8 waves; block owns MT rows; wave w owns cols [16w,16w+16)
template <bool RELU>
__global__ __launch_bounds__(256) void k_gemm_act(
    const float* __restrict__ A, const __bf16* __restrict__ Wpk,
    const float* __restrict__ bias, float* __restrict__ out)
{
  __shared__ __bf16 As[MT][136];
  const int row0 = blockIdx.x * MT;
  #pragma unroll
  for (int t = 0; t < (MT * DD) / 256; ++t) {
    int idx = threadIdx.x + t * 256;
    int r = idx >> 7, cc = idx & 127;
    As[r][cc] = (__bf16)A[(size_t)(row0 + r) * DD + cc];
  }
  __syncthreads();

  const int lane = threadIdx.x & 31;
  const int w    = threadIdx.x >> 5;
  const int m    = lane & 15;
  const int hi   = lane >> 4;
  const int n    = w * 16 + m;

  v16bf b[4];
  #pragma unroll
  for (int kk = 0; kk < 4; ++kk)   // one 32B coalesced load per lane
    b[kk] = *(const v16bf*)(Wpk + (size_t)((w * 4 + kk) * 32 + lane) * 16);

  const float bv = bias[n];
  #pragma unroll
  for (int mt = 0; mt < MT / 16; ++mt) {
    v8f c = {};
    #pragma unroll
    for (int kk = 0; kk < 4; ++kk) {
      v16bf a = load_a_frag(As + mt * 16, m, hi, kk);
      c = __builtin_amdgcn_wmma_f32_16x16x32_bf16(false, a, false, b[kk], (short)0, c,
                                                  false, false);
    }
    #pragma unroll
    for (int r = 0; r < 8; ++r) {
      int row = row0 + mt * 16 + r + 8 * hi;
      float v = c[r] + bv;
      if (RELU) v = fmaxf(v, 0.f);
      out[(size_t)row * DD + n] = v;
    }
  }
}

// out = act( (Agg * inv[row]) @ Wl + bl + H @ Wr ) — mean-div fused into tile load
template <bool RELU>
__global__ __launch_bounds__(256) void k_dual_gemm(
    const float* __restrict__ Agg, const float* __restrict__ inv,
    const float* __restrict__ H,
    const __bf16* __restrict__ Wlpk, const __bf16* __restrict__ Wrpk,
    const float* __restrict__ bias, float* __restrict__ out)
{
  __shared__ __bf16 As1[MT][136];
  __shared__ __bf16 As2[MT][136];
  const int row0 = blockIdx.x * MT;
  #pragma unroll
  for (int t = 0; t < (MT * DD) / 256; ++t) {
    int idx = threadIdx.x + t * 256;
    int r = idx >> 7, cc = idx & 127;
    float s = inv[row0 + r];
    As1[r][cc] = (__bf16)(Agg[(size_t)(row0 + r) * DD + cc] * s);
    As2[r][cc] = (__bf16)H[(size_t)(row0 + r) * DD + cc];
  }
  __syncthreads();

  const int lane = threadIdx.x & 31;
  const int w    = threadIdx.x >> 5;
  const int m    = lane & 15;
  const int hi   = lane >> 4;
  const int n    = w * 16 + m;

  v16bf bl_[4], br_[4];
  #pragma unroll
  for (int kk = 0; kk < 4; ++kk) {
    size_t o = (size_t)((w * 4 + kk) * 32 + lane) * 16;
    bl_[kk] = *(const v16bf*)(Wlpk + o);
    br_[kk] = *(const v16bf*)(Wrpk + o);
  }

  const float bv = bias[n];
  #pragma unroll
  for (int mt = 0; mt < MT / 16; ++mt) {
    v8f c = {};
    #pragma unroll
    for (int kk = 0; kk < 4; ++kk) {
      v16bf a = load_a_frag(As1 + mt * 16, m, hi, kk);
      c = __builtin_amdgcn_wmma_f32_16x16x32_bf16(false, a, false, bl_[kk], (short)0, c,
                                                  false, false);
    }
    #pragma unroll
    for (int kk = 0; kk < 4; ++kk) {
      v16bf a = load_a_frag(As2 + mt * 16, m, hi, kk);
      c = __builtin_amdgcn_wmma_f32_16x16x32_bf16(false, a, false, br_[kk], (short)0, c,
                                                  false, false);
    }
    #pragma unroll
    for (int r = 0; r < 8; ++r) {
      int row = row0 + mt * 16 + r + 8 * hi;
      float v = c[r] + bv;
      if (RELU) v = fmaxf(v, 0.f);
      out[(size_t)row * DD + n] = v;
    }
  }
}

// ---------------- host-side orchestration ----------------

extern "C" void kernel_launch(void* const* d_in, const int* in_sizes, int n_in,
                              void* d_out, int out_size, void* d_ws, size_t ws_size,
                              hipStream_t stream) {
  (void)in_sizes; (void)n_in; (void)out_size; (void)ws_size;
  const float*     x  = (const float*)d_in[0];
  const float*     Wp = (const float*)d_in[1];
  const float*     bp = (const float*)d_in[2];
  const float*     Wl = (const float*)d_in[3];
  const float*     bl = (const float*)d_in[4];
  const float*     Wr = (const float*)d_in[5];
  const long long* ei = (const long long*)d_in[6];  // [2, NE] int64
  float* out = (float*)d_out;

  char* ws = (char*)d_ws;
  size_t off = 0;
  auto alignup = [](size_t v) { return (v + 255) & ~(size_t)255; };
  float*  inv  = (float*)(ws + off);  off += alignup((size_t)NN * 4);
  float*  hp   = (float*)(ws + off);  off += alignup((size_t)NN * DD * 4);
  float*  agg  = (float*)(ws + off);  off += alignup((size_t)NN * DD * 4);
  float*  hbuf = (float*)(ws + off);  off += alignup((size_t)NN * DD * 4);
  __bf16* Wppk = (__bf16*)(ws + off); off += alignup((size_t)NL * DD * DD * 2);
  __bf16* Wlpk = (__bf16*)(ws + off); off += alignup((size_t)NL * DD * DD * 2);
  __bf16* Wrpk = (__bf16*)(ws + off); off += alignup((size_t)NL * DD * DD * 2);

  // degree -> reciprocal (clamped to 1)
  k_zero_f<<<(NN + 255) / 256, 256, 0, stream>>>(inv, NN);
  k_count<<<NE / 256, 256, 0, stream>>>(ei + NE, inv);
  k_inv<<<(NN + 255) / 256, 256, 0, stream>>>(inv);

  // weights -> bf16, pre-packed into B-fragment layout
  const int wtot = NL * DD * DD;
  k_pack_w<<<(wtot + 255) / 256, 256, 0, stream>>>(Wp, Wppk, wtot);
  k_pack_w<<<(wtot + 255) / 256, 256, 0, stream>>>(Wl, Wlpk, wtot);
  k_pack_w<<<(wtot + 255) / 256, 256, 0, stream>>>(Wr, Wrpk, wtot);

  const float* h = x;
  float* outs[NL] = { out, hbuf, out };  // ping-pong; final layer lands in d_out
  for (int l = 0; l < NL; ++l) {
    // hp = relu(h @ Wp[l] + bp[l])
    k_gemm_act<true><<<NN / MT, 256, 0, stream>>>(h, Wppk + (size_t)l * DD * DD,
                                                  bp + l * DD, hp);
    // agg = 0; agg[tgt] += hp[src]
    k_zero_f4<<<(NN * DD / 4) / 256, 256, 0, stream>>>((float4*)agg, NN * DD / 4);
    k_scatter<<<(int)(((long long)NE * 32) / 256), 256, 0, stream>>>(ei, hp, agg);
    // out = act((agg*inv) @ Wl[l] + bl[l] + h @ Wr[l])
    if (l < NL - 1)
      k_dual_gemm<true><<<NN / MT, 256, 0, stream>>>(agg, inv, h,
                                                     Wlpk + (size_t)l * DD * DD,
                                                     Wrpk + (size_t)l * DD * DD,
                                                     bl + l * DD, outs[l]);
    else
      k_dual_gemm<false><<<NN / MT, 256, 0, stream>>>(agg, inv, h,
                                                      Wlpk + (size_t)l * DD * DD,
                                                      Wrpk + (size_t)l * DD * DD,
                                                      bl + l * DD, outs[l]);
    h = outs[l];
  }
}
